// LSTM2_1812476199400
// MI455X (gfx1250) — compile-verified, matrix operand order checked by reading, not run on hardware
//
#include <hip/hip_runtime.h>
#include <math.h>

typedef __attribute__((ext_vector_type(16))) _Float16 v16h;
typedef __attribute__((ext_vector_type(8)))  float    v8f;

#define B_   1024
#define T_   512
#define F_   16
#define H_   12
#define HP_  16     // padded hidden stride (bytes*2 = 32B rows, async-B128 friendly)
#define L_   5
#define OUT_ 8
#define FC_  128

#define LOG2E  1.44269504088896340736f

// fast sigmoid: v_exp_f32 + v_rcp_f32
__device__ __forceinline__ float fsig(float x) {
    return __builtin_amdgcn_rcpf(1.0f + __builtin_amdgcn_exp2f(-LOG2E * x));
}
// fast tanh: prefer gfx1250 v_tanh_f32
__device__ __forceinline__ float ftanh(float x) {
#if __has_builtin(__builtin_amdgcn_tanhf)
    return __builtin_amdgcn_tanhf(x);
#elif __has_builtin(__builtin_amdgcn_tanh_f32)
    return __builtin_amdgcn_tanh_f32(x);
#else
    // tanh(x) = 1 - 2/(exp(2x)+1)
    float e = __builtin_amdgcn_exp2f(2.0f * LOG2E * x);
    return 1.0f - 2.0f * __builtin_amdgcn_rcpf(e + 1.0f);
#endif
}

// ---------------------------------------------------------------------------
// Convert x f32 -> f16 once (stride F_==HP_ already).
// ---------------------------------------------------------------------------
__global__ __launch_bounds__(256) void xcvt_kernel(const float* __restrict__ x,
                                                   _Float16* __restrict__ x16) {
    const size_t n = (size_t)B_ * T_ * F_;
    for (size_t i = (size_t)blockIdx.x * blockDim.x + threadIdx.x; i < n;
         i += (size_t)gridDim.x * blockDim.x)
        x16[i] = (_Float16)x[i];
}

// ---------------------------------------------------------------------------
// Pack weights into WMMA-friendly padded f16 layouts (runs once, tiny).
// ---------------------------------------------------------------------------
__global__ __launch_bounds__(256) void pack_kernel(
    const float* __restrict__ W_ih0, const float* __restrict__ W_hh0,
    const float* __restrict__ b0,    const float* __restrict__ W_ih_r,
    const float* __restrict__ W_hh_r,const float* __restrict__ b_r,
    const float* __restrict__ fc1_w, const float* __restrict__ fc1_b,
    const float* __restrict__ fc2_w, const float* __restrict__ fc2_b,
    _Float16* __restrict__ Wp, float* __restrict__ Bp,
    _Float16* __restrict__ W1p, float* __restrict__ b1p,
    _Float16* __restrict__ W2p, float* __restrict__ b2p)
{
    const int tid = threadIdx.x;
    // LSTM weights: Wp[l][n=gate*16+j][k], k<16 x-part, k in [16,28) h-part
    for (int idx = tid; idx < L_ * 64 * 32; idx += 256) {
        int l = idx / (64 * 32);
        int n = (idx / 32) & 63;
        int k = idx & 31;
        int gate = n >> 4, j = n & 15;
        float v = 0.0f;
        if (j < H_) {
            int row = gate * H_ + j;
            if (l == 0) {
                if (k < 16)                      v = W_ih0[row * F_ + k];
                else if (k - 16 < H_)            v = W_hh0[row * H_ + (k - 16)];
            } else {
                const float* wi = W_ih_r + (size_t)(l - 1) * 4 * H_ * H_;
                const float* wh = W_hh_r + (size_t)(l - 1) * 4 * H_ * H_;
                if (k < H_)                      v = wi[row * H_ + k];
                else if (k >= 16 && k - 16 < H_) v = wh[row * H_ + (k - 16)];
            }
        }
        Wp[idx] = (_Float16)v;
    }
    // LSTM biases (gate-padded)
    for (int idx = tid; idx < L_ * 64; idx += 256) {
        int l = idx / 64, n = idx & 63;
        int gate = n >> 4, j = n & 15;
        float v = 0.0f;
        if (j < H_) v = (l == 0) ? b0[gate * H_ + j] : b_r[(l - 1) * 4 * H_ + gate * H_ + j];
        Bp[idx] = v;
    }
    // fc1 (K: 60 -> 64)
    for (int idx = tid; idx < FC_ * 64; idx += 256) {
        int n = idx >> 6, k = idx & 63;
        W1p[idx] = (_Float16)((k < L_ * H_) ? fc1_w[n * (L_ * H_) + k] : 0.0f);
    }
    for (int idx = tid; idx < FC_; idx += 256) b1p[idx] = fc1_b[idx];
    // fc2 (N: 8 -> 16)
    for (int idx = tid; idx < 16 * FC_; idx += 256) {
        int n = idx >> 7, k = idx & 127;
        W2p[idx] = (_Float16)((n < OUT_) ? fc2_w[n * FC_ + k] : 0.0f);
    }
    for (int idx = tid; idx < 16; idx += 256) b2p[idx] = (idx < OUT_) ? fc2_b[idx] : 0.0f;
}

// ---------------------------------------------------------------------------
// LSTM: one workgroup = 16 batch rows, all 5 layers, all 512 timesteps.
// Input staging via async global->LDS (wave 0, B128 per lane); 4 waves each
// run one v_wmma_f32_16x16x32_f16 per timestep (one gate tile, bias in C).
// All inter-layer buffers are f16 with padded stride HP_=16.
// ---------------------------------------------------------------------------
__global__ __launch_bounds__(128) void lstm_kernel(
    const _Float16* __restrict__ x16,  // [B,T,16] f16
    const _Float16* __restrict__ Wp,   // [L][64][32]
    const float* __restrict__ Bp,      // [L][64]
    _Float16* __restrict__ bufA,       // [B,T,HP] ping
    _Float16* __restrict__ bufB)       // [B,T,HP] pong
{
    __shared__ __align__(16) _Float16 Ah[16][32];   // [row][ x(0..15) | h(16..31) ]
    __shared__ float G[4][16][16];                  // gate pre-activations
    __shared__ float Cst[16][H_];                   // cell state

    const int tid  = threadIdx.x;
    const int wave = tid >> 5;         // 4 waves = 4 gate tiles
    const int ln   = tid & 31;
    const int M    = ln & 15;
    const int kb   = (ln < 16) ? 0 : 8;
    const int Mr   = (ln < 16) ? 0 : 8;
    const int brow = blockIdx.x * 16;

    // async staging geometry: lane -> (row, 16B chunk) of the 16x16-half x tile
    const int s_r = ln >> 1;
    const int s_c = (ln & 1) * 8;      // half offset within row
    const unsigned ldsoff = (unsigned)(uintptr_t)&Ah[s_r][s_c];

    for (int l = 0; l < L_; ++l) {
        const _Float16* inb = (l == 0) ? x16 : ((l & 1) ? bufA : bufB);
        _Float16*      outb = (l & 1) ? bufB : bufA;

        // zero LDS state for this layer
        for (int idx = tid; idx < 16 * 32; idx += 128) Ah[idx >> 5][idx & 31] = (_Float16)0.0f;
        for (int idx = tid; idx < 16 * H_; idx += 128) Cst[idx / H_][idx % H_] = 0.0f;

        // B fragment: constant over T, one per wave (gate)
        v16h bfrag;
        const _Float16* wl = Wp + ((size_t)l * 64 + wave * 16 + M) * 32;
        #pragma unroll
        for (int i = 0; i < 8; ++i) { bfrag[i] = wl[kb + i]; bfrag[8 + i] = wl[16 + kb + i]; }
        const float bias = Bp[l * 64 + wave * 16 + M];
        const _Float16* srow = inb + ((size_t)(brow + s_r) * T_) * HP_ + s_c;
        __syncthreads();

        for (int t = 0; t < T_; ++t) {
            // async-stage x_t tile (16 rows x 16 halves) into Ah cols [0,16)
            if (wave == 0) {
                unsigned long long ga = (unsigned long long)(uintptr_t)(srow + (size_t)t * HP_);
                asm volatile("global_load_async_to_lds_b128 %0, %1, off"
                             :: "v"(ldsoff), "v"(ga) : "memory");
            }
            asm volatile("s_wait_asynccnt 0x0" ::: "memory");
            __syncthreads();

            // gather A fragment (ISA 16-bit A 16x32 layout)
            v16h afrag;
            #pragma unroll
            for (int i = 0; i < 8; ++i) { afrag[i] = Ah[M][kb + i]; afrag[8 + i] = Ah[M][16 + kb + i]; }

            v8f acc;
            #pragma unroll
            for (int vv = 0; vv < 8; ++vv) acc[vv] = bias;
            acc = __builtin_amdgcn_wmma_f32_16x16x32_f16(
                false, afrag, false, bfrag, (short)0, acc, false, false);

            #pragma unroll
            for (int vv = 0; vv < 8; ++vv) G[wave][Mr + vv][M] = acc[vv];
            __syncthreads();

            // elementwise LSTM update (fast hw transcendentals), 16x16 padded
            for (int idx = tid; idx < 16 * HP_; idx += 128) {
                int r = idx >> 4, j = idx & 15;
                size_t o = ((size_t)(brow + r) * T_ + t) * HP_ + j;
                if (j < H_) {
                    float ig = fsig(G[0][r][j]);
                    float fg = fsig(G[1][r][j]);
                    float gg = ftanh(G[2][r][j]);
                    float og = fsig(G[3][r][j]);
                    float c  = fg * Cst[r][j] + ig * gg;
                    float h  = og * ftanh(c);
                    Cst[r][j] = c;
                    _Float16 h16 = (_Float16)h;
                    Ah[r][16 + j] = h16;       // h_t for next step
                    outb[o] = h16;
                } else {
                    outb[o] = (_Float16)0.0f;  // keep pad columns exactly zero
                }
            }
            __syncthreads();
        }
        __syncthreads();
    }
}

// ---------------------------------------------------------------------------
// Head: z = relu(flatten last-5 hidden); fc1 (2 WMMAs x 8 waves); relu;
// fc2 (4 WMMAs on wave 0); +bias; f32 out.
// ---------------------------------------------------------------------------
__global__ __launch_bounds__(256) void head_kernel(
    const _Float16* __restrict__ hid,  // [B,T,HP] (last layer = bufA)
    const _Float16* __restrict__ W1p,  // [128][64]
    const float*    __restrict__ b1p,  // [128]
    const _Float16* __restrict__ W2p,  // [16][128]
    const float*    __restrict__ b2p,  // [16]
    float* __restrict__ out)           // [B,8]
{
    __shared__ _Float16 Az[16][64];
    __shared__ _Float16 Z1[16][FC_];

    const int tid  = threadIdx.x;
    const int wave = tid >> 5;
    const int ln   = tid & 31;
    const int M    = ln & 15;
    const int kb   = (ln < 16) ? 0 : 8;
    const int Mr   = (ln < 16) ? 0 : 8;
    const int brow = blockIdx.x * 16;

    // build relu(z) tile, K padded 60->64
    for (int idx = tid; idx < 16 * 64; idx += 256) {
        int r = idx >> 6, k = idx & 63;
        float v = 0.0f;
        if (k < L_ * H_) {
            int ts = k / H_, j = k % H_;
            v = (float)hid[((size_t)(brow + r) * T_ + (T_ - L_ + ts)) * HP_ + j];
            v = v > 0.0f ? v : 0.0f;
        }
        Az[r][k] = (_Float16)v;
    }
    __syncthreads();

    // fc1: wave w -> columns [16w, 16w+16), two K=32 WMMAs
    v16h a0, a1, w0, w1;
    const _Float16* wrow = W1p + (size_t)(wave * 16 + M) * 64;
    #pragma unroll
    for (int i = 0; i < 8; ++i) {
        a0[i] = Az[M][kb + i];        a0[8 + i] = Az[M][16 + kb + i];
        a1[i] = Az[M][32 + kb + i];   a1[8 + i] = Az[M][48 + kb + i];
        w0[i] = wrow[kb + i];         w0[8 + i] = wrow[16 + kb + i];
        w1[i] = wrow[32 + kb + i];    w1[8 + i] = wrow[48 + kb + i];
    }
    v8f acc;
    const float bias1 = b1p[wave * 16 + M];
    #pragma unroll
    for (int vv = 0; vv < 8; ++vv) acc[vv] = bias1;
    acc = __builtin_amdgcn_wmma_f32_16x16x32_f16(false, a0, false, w0, (short)0, acc, false, false);
    acc = __builtin_amdgcn_wmma_f32_16x16x32_f16(false, a1, false, w1, (short)0, acc, false, false);
    #pragma unroll
    for (int vv = 0; vv < 8; ++vv) {
        float z = acc[vv]; z = z > 0.0f ? z : 0.0f;
        Z1[Mr + vv][wave * 16 + M] = (_Float16)z;
    }
    __syncthreads();

    // fc2 on wave 0 only (uniform within the wave -> EXEC all-ones)
    if (wave == 0) {
        v8f acc2;
        const float bias2 = b2p[M];
        #pragma unroll
        for (int vv = 0; vv < 8; ++vv) acc2[vv] = bias2;
        #pragma unroll
        for (int kt = 0; kt < 4; ++kt) {
            v16h af, bf;
            const _Float16* w2row = W2p + (size_t)M * FC_ + kt * 32;
            #pragma unroll
            for (int i = 0; i < 8; ++i) {
                af[i]     = Z1[M][kt * 32 + kb + i];
                af[8 + i] = Z1[M][kt * 32 + 16 + kb + i];
                bf[i]     = w2row[kb + i];
                bf[8 + i] = w2row[16 + kb + i];
            }
            acc2 = __builtin_amdgcn_wmma_f32_16x16x32_f16(false, af, false, bf, (short)0, acc2, false, false);
        }
        if (M < OUT_) {
            #pragma unroll
            for (int vv = 0; vv < 8; ++vv)
                out[(size_t)(brow + Mr + vv) * OUT_ + M] = acc2[vv];
        }
    }
}

// ---------------------------------------------------------------------------
extern "C" void kernel_launch(void* const* d_in, const int* in_sizes, int n_in,
                              void* d_out, int out_size, void* d_ws, size_t ws_size,
                              hipStream_t stream) {
    const float* x      = (const float*)d_in[0];
    const float* W_ih0  = (const float*)d_in[1];
    const float* W_hh0  = (const float*)d_in[2];
    const float* b0     = (const float*)d_in[3];
    const float* W_ih_r = (const float*)d_in[4];
    const float* W_hh_r = (const float*)d_in[5];
    const float* b_r    = (const float*)d_in[6];
    const float* fc1_w  = (const float*)d_in[7];
    const float* fc1_b  = (const float*)d_in[8];
    const float* fc2_w  = (const float*)d_in[9];
    const float* fc2_b  = (const float*)d_in[10];
    float* out = (float*)d_out;

    // workspace carve-up (256B aligned)
    char* ws = (char*)d_ws;
    size_t off = 0;
    auto alloc = [&](size_t bytes) { void* p = ws + off; off = (off + bytes + 255) & ~(size_t)255; return p; };
    _Float16* Wp   = (_Float16*)alloc((size_t)L_ * 64 * 32 * sizeof(_Float16));
    float*    Bp   = (float*)   alloc((size_t)L_ * 64 * sizeof(float));
    _Float16* W1p  = (_Float16*)alloc((size_t)FC_ * 64 * sizeof(_Float16));
    float*    b1p  = (float*)   alloc((size_t)FC_ * sizeof(float));
    _Float16* W2p  = (_Float16*)alloc((size_t)16 * FC_ * sizeof(_Float16));
    float*    b2p  = (float*)   alloc((size_t)16 * sizeof(float));
    _Float16* x16  = (_Float16*)alloc((size_t)B_ * T_ * F_  * sizeof(_Float16));
    _Float16* bufA = (_Float16*)alloc((size_t)B_ * T_ * HP_ * sizeof(_Float16));
    _Float16* bufB = (_Float16*)alloc((size_t)B_ * T_ * HP_ * sizeof(_Float16));
    (void)ws_size; (void)in_sizes; (void)n_in; (void)out_size;

    xcvt_kernel<<<4096, 256, 0, stream>>>(x, x16);
    pack_kernel<<<1, 256, 0, stream>>>(W_ih0, W_hh0, b0, W_ih_r, W_hh_r, b_r,
                                       fc1_w, fc1_b, fc2_w, fc2_b,
                                       Wp, Bp, W1p, b1p, W2p, b2p);
    lstm_kernel<<<B_ / 16, 128, 0, stream>>>(x16, Wp, Bp, bufA, bufB);
    // layers 0,2,4 write bufA -> final layer output lives in bufA
    head_kernel<<<B_ / 16, 256, 0, stream>>>(bufA, W1p, b1p, W2p, b2p, out);
}
